// FLOT_56831007261015
// MI455X (gfx1250) — matrix operand order repaired and checked by reading, not compiled, not run on hardware
//
#include <hip/hip_runtime.h>
#include <hip/hip_bf16.h>

// ---------------------------------------------------------------------------
// FLOT scene-flow forward for MI455X (gfx1250, wave32, WMMA).
// GEMM-shaped work (set-conv 1x1 convs, sinkhorn feature correlation) runs on
// v_wmma_f32_16x16x32_f16. The 4096x4096 transport matrix is never stored:
// sinkhorn (1 iter) is reformulated as two WMMA passes with exp() epilogues.
// Round 3: GEMM templated on k-fragment count (kdim in {32,64,96,128}) so the
// inner loop is branch-free straight-line load/wmma code; weight fragments
// stay hoisted across 4 row tiles per wave.
// ---------------------------------------------------------------------------

typedef __attribute__((ext_vector_type(16))) _Float16 v16h;
typedef __attribute__((ext_vector_type(8)))  float    v8f;

#define B_SZ   2
#define N_PTS  4096
#define K_NN   32
#define LDH    128                 // fixed f16 row stride (256B) for all activation buffers
#define NKROWS (N_PTS * K_NN)      // rows per batch in conv GEMMs
#define RTOT   (B_SZ * NKROWS)     // total conv GEMM rows

// ---------------- WMMA fragment helpers (16x16x32 f16 layout) --------------
// A (16x32): lane<16 holds row lane, K {0..7,16..23}; lane>=16 same rows, K {8..15,24..31}
// B (32x16): lane holds column lane&15; lanes<16 K {0..7,16..23}, lanes>=16 K {8..15,24..31} (same as A)
__device__ __forceinline__ v16h load_a_frag(const _Float16* rowptr, int k0, int sel) {
  union { uint4 u[2]; v16h v; } h;
  h.u[0] = *reinterpret_cast<const uint4*>(rowptr + k0 + sel * 8);
  h.u[1] = *reinterpret_cast<const uint4*>(rowptr + k0 + sel * 8 + 16);
  return h.v;
}
__device__ __forceinline__ v16h load_b_frag(const _Float16* rowptr, int k0, int sel) {
  union { uint4 u[2]; v16h v; } h;
  h.u[0] = *reinterpret_cast<const uint4*>(rowptr + k0 + sel * 8);
  h.u[1] = *reinterpret_cast<const uint4*>(rowptr + k0 + sel * 8 + 16);
  return h.v;
}
__device__ __forceinline__ v8f wmma_fma(v16h a, v16h b, v8f c) {
  return __builtin_amdgcn_wmma_f32_16x16x32_f16(false, a, false, b, (short)0, c, false, false);
}

// ---------------- scalars: eps, gamma, power, 1/eps ------------------------
__global__ void flot_scalars(const float* epsP, const float* gamP, float* s) {
  if (threadIdx.x == 0 && blockIdx.x == 0) {
    float eps = __expf(epsP[0]) + 0.03f;
    float gam = __expf(gamP[0]);
    s[0] = eps; s[1] = gam; s[2] = gam / (gam + eps); s[3] = 1.0f / eps;
  }
}

// ---------------- |p|^2 ----------------------------------------------------
__global__ void flot_sqnorm(const float* pc, float* sq, int total) {
  int i = blockIdx.x * blockDim.x + threadIdx.x;
  if (i >= total) return;
  float x = pc[i * 3 + 0], y = pc[i * 3 + 1], z = pc[i * 3 + 2];
  sq[i] = x * x + y * y + z * z;
}

// ---------------- KNN (top-32 smallest dist, incl self) --------------------
__global__ __launch_bounds__(256) void flot_knn(const float* pc, int* nbr, int N) {
  int b = blockIdx.y;
  int n = blockIdx.x * blockDim.x + threadIdx.x;
  const float* base = pc + (size_t)b * N * 3;
  float px = base[n * 3 + 0], py = base[n * 3 + 1], pz = base[n * 3 + 2];
  float bd[K_NN]; int bi[K_NN];
  #pragma unroll
  for (int k = 0; k < K_NN; ++k) { bd[k] = 3.0e38f; bi[k] = 0; }
  __shared__ float sx[256], sy[256], sz[256];
  for (int t0 = 0; t0 < N; t0 += 256) {
    int m = t0 + threadIdx.x;
    sx[threadIdx.x] = base[m * 3 + 0];
    sy[threadIdx.x] = base[m * 3 + 1];
    sz[threadIdx.x] = base[m * 3 + 2];
    __syncthreads();
    for (int j = 0; j < 256; ++j) {
      float dx = px - sx[j], dy = py - sy[j], dz = pz - sz[j];
      float d = dx * dx + dy * dy + dz * dz;
      if (d < bd[K_NN - 1]) {
        int pos = K_NN - 1;
        while (pos > 0 && bd[pos - 1] > d) {
          bd[pos] = bd[pos - 1]; bi[pos] = bi[pos - 1]; --pos;
        }
        bd[pos] = d; bi[pos] = t0 + j;
      }
    }
    __syncthreads();
  }
  int* out = nbr + ((size_t)b * N + n) * K_NN;
  for (int k = 0; k < K_NN; ++k) out[k] = bi[k];
}

// ---------------- gather + edge feats + pad -> f16 rows --------------------
__global__ void flot_gather(const float* x, int xstride, int cin, const float* pc,
                            const int* nbr, _Float16* h, int cinpad, int N) {
  size_t r = (size_t)blockIdx.x * blockDim.x + threadIdx.x;
  if (r >= (size_t)RTOT) return;
  int b = (int)(r / NKROWS);
  int rem = (int)(r % NKROWS);
  int n = rem / K_NN;
  int m = nbr[r];
  const float* xs = x + ((size_t)b * N + m) * xstride;
  const float* pm = pc + ((size_t)b * N + m) * 3;
  const float* pn = pc + ((size_t)b * N + n) * 3;
  _Float16* hr = h + r * LDH;
  for (int c = 0; c < cin; ++c) hr[c] = (_Float16)xs[c];
  hr[cin + 0] = (_Float16)(pm[0] - pn[0]);
  hr[cin + 1] = (_Float16)(pm[1] - pn[1]);
  hr[cin + 2] = (_Float16)(pm[2] - pn[2]);
  for (int c = cin + 3; c < cinpad; ++c) hr[c] = (_Float16)0.0f;
}

// ---------------- weight repack: f32 (cout,cin) -> f16 (cout,cinpad) -------
__global__ void flot_repack_w(const float* W, _Float16* Wh, int cout, int cin, int cinpad) {
  int i = blockIdx.x * blockDim.x + threadIdx.x;
  if (i >= cout * cinpad) return;
  int o = i / cinpad, c = i % cinpad;
  Wh[i] = (_Float16)((c < cin) ? W[o * cin + c] : 0.0f);
}

// ---------------- WMMA GEMM: out[r,o] = sum_i A[r,i] * W[o,i] --------------
// KF = kdim/32 (compile-time). Each wave handles 4 consecutive row tiles of
// one column tile; all KF weight fragments are preloaded once and reused.
template <int KF>
__global__ __launch_bounds__(256) void flot_gemm_wmma(const _Float16* __restrict__ A,
                                                      const _Float16* __restrict__ W,
                                                      _Float16* __restrict__ Cc,
                                                      int rows, int cout) {
  const int kdim = KF * 32;
  int wave = (int)((blockIdx.x * (size_t)blockDim.x + threadIdx.x) >> 5);
  int lane = threadIdx.x & 31;
  int colTiles = cout >> 4;
  int superRows = rows >> 6;           // 4 row tiles per wave
  if (wave >= superRows * colTiles) return;
  int sr = wave / colTiles, ct = wave % colTiles;
  int nn = lane & 15, sel = lane >> 4;
  const _Float16* wrow = W + (size_t)(ct * 16 + nn) * kdim;
  v16h wf[KF];
  #pragma unroll
  for (int kf = 0; kf < KF; ++kf) wf[kf] = load_b_frag(wrow, kf * 32, sel);
  int col = ct * 16 + nn;
  #pragma unroll
  for (int rr = 0; rr < 4; ++rr) {
    int r0 = sr * 64 + rr * 16;
    const _Float16* arow = A + (size_t)(r0 + nn) * LDH;
    v8f acc = {};
    #pragma unroll
    for (int kf = 0; kf < KF; ++kf)
      acc = wmma_fma(load_a_frag(arow, kf * 32, sel), wf[kf], acc);
    int mbase = r0 + (sel ? 8 : 0);
    #pragma unroll
    for (int v = 0; v < 8; ++v)
      Cc[(size_t)(mbase + v) * LDH + col] = (_Float16)acc[v];
  }
}

// ---------------- InstanceNorm stats (one block per (b,c)) -----------------
__global__ __launch_bounds__(256) void flot_in_stats(const _Float16* y, float* stats, int C) {
  int b = blockIdx.x / C, c = blockIdx.x % C;
  const _Float16* p = y + (size_t)b * NKROWS * LDH + c;
  float s = 0.f, s2 = 0.f;
  for (int i = threadIdx.x; i < NKROWS; i += 256) {
    float v = (float)p[(size_t)i * LDH];
    s += v; s2 += v * v;
  }
  __shared__ float rs[256], rs2[256];
  rs[threadIdx.x] = s; rs2[threadIdx.x] = s2;
  __syncthreads();
  for (int st = 128; st > 0; st >>= 1) {
    if (threadIdx.x < st) { rs[threadIdx.x] += rs[threadIdx.x + st]; rs2[threadIdx.x] += rs2[threadIdx.x + st]; }
    __syncthreads();
  }
  if (threadIdx.x == 0) {
    float mu = rs[0] / (float)NKROWS;
    float var = rs2[0] / (float)NKROWS - mu * mu;
    stats[(b * C + c) * 2 + 0] = mu;
    stats[(b * C + c) * 2 + 1] = rsqrtf(var + 1e-5f);
  }
}

// ---------------- InstanceNorm apply + LeakyReLU(0.1), in place ------------
__global__ void flot_in_apply(_Float16* y, const float* stats, const float* g,
                              const float* bb, int C) {
  size_t idx = (size_t)blockIdx.x * blockDim.x + threadIdx.x;
  size_t tot = (size_t)RTOT * C;
  if (idx >= tot) return;
  int c = (int)(idx % C);
  size_t r = idx / C;
  int b = (int)(r / NKROWS);
  float mu = stats[(b * C + c) * 2 + 0];
  float rsd = stats[(b * C + c) * 2 + 1];
  float v = ((float)y[r * LDH + c] - mu) * rsd * g[c] + bb[c];
  v = (v >= 0.f) ? v : 0.1f * v;
  y[r * LDH + c] = (_Float16)v;
}

// ---------------- max-pool over K neighbors -> f32 feats -------------------
__global__ void flot_maxpool(const _Float16* y, float* feats, int C, int N) {
  int idx = blockIdx.x * blockDim.x + threadIdx.x;
  if (idx >= B_SZ * N * C) return;
  int c = idx % C;
  int pnt = idx / C;          // b*N + n
  const _Float16* p = y + ((size_t)pnt * K_NN) * LDH + c;
  float m = -3.0e38f;
  #pragma unroll
  for (int k = 0; k < K_NN; ++k) {
    float v = (float)p[(size_t)k * LDH];
    m = (v > m) ? v : m;
  }
  feats[(size_t)pnt * LDH + c] = m;
}

// ---------------- feature L2-normalize -> f16 ------------------------------
__global__ __launch_bounds__(128) void flot_featnorm(const float* f, _Float16* out) {
  int p = blockIdx.x;                     // b*N + n
  int c = threadIdx.x;                    // 128 channels
  float v = f[(size_t)p * LDH + c];
  __shared__ float red[128];
  red[c] = v * v;
  __syncthreads();
  for (int st = 64; st > 0; st >>= 1) {
    if (c < st) red[c] += red[c + st];
    __syncthreads();
  }
  float rs = rsqrtf(red[0] + 1e-8f);
  out[(size_t)p * LDH + c] = (_Float16)(v * rs);
}

// ---------------- sinkhorn pass 1: column sums of K (WMMA + exp) -----------
// One block owns a 16-column strip; B fragments are loop-invariant -> hoisted.
__global__ __launch_bounds__(128) void flot_colsum(const _Float16* f1, const _Float16* f2,
                                                   const float* pc1, const float* pc2,
                                                   const float* sq1, const float* sq2,
                                                   const float* scal, float* colsum, int N) {
  int tilesM = N >> 4;
  int b = blockIdx.x / tilesM;
  int mt = blockIdx.x % tilesM;
  int wave = threadIdx.x >> 5, lane = threadIdx.x & 31;
  int nn = lane & 15, sel = lane >> 4;
  __shared__ float csum[16];
  if (threadIdx.x < 16) csum[threadIdx.x] = 0.f;
  __syncthreads();
  float inv_eps = scal[3];
  int m0 = mt * 16;
  int cm = b * N + m0 + nn;
  const _Float16* brow = f2 + (size_t)cm * LDH;
  v16h bf[4];
  #pragma unroll
  for (int kf = 0; kf < 4; ++kf) bf[kf] = load_b_frag(brow, kf * 32, sel);
  float p2x = pc2[cm * 3 + 0], p2y = pc2[cm * 3 + 1], p2z = pc2[cm * 3 + 2];
  float s2 = sq2[cm];
  float partial = 0.f;
  for (int rt = wave; rt < (N >> 4); rt += 4) {
    int r0 = rt * 16;
    const _Float16* arow = f1 + (size_t)(b * N + r0 + nn) * LDH;
    v8f acc = {};
    #pragma unroll
    for (int kf = 0; kf < 4; ++kf)
      acc = wmma_fma(load_a_frag(arow, kf * 32, sel), bf[kf], acc);
    int mbase = r0 + (sel ? 8 : 0);
    #pragma unroll
    for (int v = 0; v < 8; ++v) {
      int rn = b * N + mbase + v;
      float geom = sq1[rn] + s2 - 2.f * (pc1[rn * 3] * p2x + pc1[rn * 3 + 1] * p2y + pc1[rn * 3 + 2] * p2z);
      float Kv = (geom < 100.f) ? __expf(-(1.f - acc[v]) * inv_eps) : 0.f;
      partial += Kv;
    }
  }
  atomicAdd(&csum[nn], partial);
  __syncthreads();
  if (threadIdx.x < 16) colsum[b * N + m0 + threadIdx.x] = csum[threadIdx.x];
}

// ---------------- b vector -------------------------------------------------
__global__ void flot_bvec(const float* colsum, const float* scal, float* bvec, int N) {
  int i = blockIdx.x * blockDim.x + threadIdx.x;
  if (i >= B_SZ * N) return;
  float KTa = colsum[i] * (1.0f / (float)N);
  bvec[i] = powf((1.0f / (float)N) / (KTa + 1e-8f), scal[2]);
}

// ---------------- sinkhorn pass 2: row sums of K*b and K*b*pc2 -------------
// One block owns a 16-row strip; A fragments are loop-invariant -> hoisted.
__global__ __launch_bounds__(128) void flot_rowpass(const _Float16* f1, const _Float16* f2,
                                                    const float* pc1, const float* pc2,
                                                    const float* sq1, const float* sq2,
                                                    const float* bvec, const float* scal,
                                                    float* rowacc, int N) {
  int tilesN = N >> 4;
  int b = blockIdx.x / tilesN;
  int nt = blockIdx.x % tilesN;
  int wave = threadIdx.x >> 5, lane = threadIdx.x & 31;
  int nn = lane & 15, sel = lane >> 4;
  __shared__ float acc4[16][4];
  if (threadIdx.x < 64) acc4[threadIdx.x >> 2][threadIdx.x & 3] = 0.f;
  __syncthreads();
  float inv_eps = scal[3];
  int r0 = nt * 16;
  const _Float16* arow = f1 + (size_t)(b * N + r0 + nn) * LDH;
  v16h af[4];
  #pragma unroll
  for (int kf = 0; kf < 4; ++kf) af[kf] = load_a_frag(arow, kf * 32, sel);
  int mbase = (sel ? 8 : 0);
  float rpx[8], rpy[8], rpz[8], rsq[8];
  #pragma unroll
  for (int v = 0; v < 8; ++v) {
    int rn = b * N + r0 + mbase + v;
    rpx[v] = pc1[rn * 3 + 0]; rpy[v] = pc1[rn * 3 + 1]; rpz[v] = pc1[rn * 3 + 2];
    rsq[v] = sq1[rn];
  }
  for (int ct = wave; ct < (N >> 4); ct += 4) {
    int cm = b * N + ct * 16 + nn;
    const _Float16* brow = f2 + (size_t)cm * LDH;
    float p2x = pc2[cm * 3 + 0], p2y = pc2[cm * 3 + 1], p2z = pc2[cm * 3 + 2];
    float s2 = sq2[cm];
    float bm = bvec[cm];
    v8f acc = {};
    #pragma unroll
    for (int kf = 0; kf < 4; ++kf)
      acc = wmma_fma(af[kf], load_b_frag(brow, kf * 32, sel), acc);
    #pragma unroll
    for (int v = 0; v < 8; ++v) {
      float geom = rsq[v] + s2 - 2.f * (rpx[v] * p2x + rpy[v] * p2y + rpz[v] * p2z);
      float Kv = (geom < 100.f) ? __expf(-(1.f - acc[v]) * inv_eps) : 0.f;
      float w = Kv * bm;
      int ml = mbase + v;
      atomicAdd(&acc4[ml][0], w);
      atomicAdd(&acc4[ml][1], w * p2x);
      atomicAdd(&acc4[ml][2], w * p2y);
      atomicAdd(&acc4[ml][3], w * p2z);
    }
  }
  __syncthreads();
  if (threadIdx.x < 64) {
    int t = threadIdx.x >> 2, j = threadIdx.x & 3;
    rowacc[((size_t)b * N + r0 + t) * 4 + j] = acc4[t][j];
  }
}

// ---------------- OT flow --------------------------------------------------
__global__ void flot_otflow(const float* rowacc, const float* scal, const float* pc1,
                            float* otflow, int N) {
  int i = blockIdx.x * blockDim.x + threadIdx.x;
  if (i >= B_SZ * N) return;
  float s0 = rowacc[(size_t)i * 4];
  float a = powf((1.0f / (float)N) / (s0 + 1e-8f), scal[2]);
  float den = a * s0 + 1e-8f;
  #pragma unroll
  for (int c = 0; c < 3; ++c)
    otflow[i * 3 + c] = a * rowacc[(size_t)i * 4 + 1 + c] / den - pc1[i * 3 + c];
}

// ---------------- final FC + residual --------------------------------------
__global__ void flot_fc(const float* feats, const float* fcw, const float* fcb,
                        const float* otflow, float* out, int total) {
  int i = blockIdx.x * blockDim.x + threadIdx.x;
  if (i >= total) return;
  float a0 = fcb[0], a1 = fcb[1], a2 = fcb[2];
  const float* f = feats + (size_t)i * LDH;
  for (int c = 0; c < 128; ++c) {
    float v = f[c];
    a0 += v * fcw[0 * 128 + c];
    a1 += v * fcw[1 * 128 + c];
    a2 += v * fcw[2 * 128 + c];
  }
  out[i * 3 + 0] = otflow[i * 3 + 0] + a0;
  out[i * 3 + 1] = otflow[i * 3 + 1] + a1;
  out[i * 3 + 2] = otflow[i * 3 + 2] + a2;
}

// ===========================================================================
extern "C" void kernel_launch(void* const* d_in, const int* in_sizes, int n_in,
                              void* d_out, int out_size, void* d_ws, size_t ws_size,
                              hipStream_t stream) {
  (void)in_sizes; (void)n_in; (void)out_size; (void)ws_size;
  const int N = N_PTS;

  // Input leaf order: pc1, pc2, then params leaves alphabetically:
  // epsilon, fc_b, fc_w, feat[0..2]{W0,W1,W2,b0,b1,b2,g0,g1,g2}, gamma, ref[0..2]{...}
  const float* pc1  = (const float*)d_in[0];
  const float* pc2  = (const float*)d_in[1];
  const float* epsP = (const float*)d_in[2];
  const float* fcb  = (const float*)d_in[3];
  const float* fcw  = (const float*)d_in[4];
  const int featBase = 5;
  const float* gamP = (const float*)d_in[32];
  const int refBase = 33;

  // ---- workspace carve-out (256B aligned) ----
  char* wsb = (char*)d_ws;
  size_t off = 0;
  auto alloc = [&](size_t bytes) -> void* {
    off = (off + 255) & ~(size_t)255;
    void* p = wsb + off;
    off += bytes;
    return p;
  };
  _Float16* h_buf    = (_Float16*)alloc((size_t)RTOT * LDH * 2);
  _Float16* conv_buf = (_Float16*)alloc((size_t)RTOT * LDH * 2);
  _Float16* fnorm1   = (_Float16*)alloc((size_t)B_SZ * N * LDH * 2);
  _Float16* fnorm2   = (_Float16*)alloc((size_t)B_SZ * N * LDH * 2);
  float*    feats_a  = (float*)alloc((size_t)B_SZ * N * LDH * 4);
  float*    feats_b  = (float*)alloc((size_t)B_SZ * N * LDH * 4);
  int*      nbr1     = (int*)alloc((size_t)B_SZ * N * K_NN * 4);
  int*      nbr2     = (int*)alloc((size_t)B_SZ * N * K_NN * 4);
  _Float16* wbuf     = (_Float16*)alloc(128 * 128 * 2);
  float*    stats    = (float*)alloc((size_t)B_SZ * 128 * 2 * 4);
  float*    sq1      = (float*)alloc((size_t)B_SZ * N * 4);
  float*    sq2      = (float*)alloc((size_t)B_SZ * N * 4);
  float*    colsum   = (float*)alloc((size_t)B_SZ * N * 4);
  float*    bvec     = (float*)alloc((size_t)B_SZ * N * 4);
  float*    rowacc   = (float*)alloc((size_t)B_SZ * N * 4 * 4);
  float*    otflow   = (float*)alloc((size_t)B_SZ * N * 3 * 4);
  float*    scalars  = (float*)alloc(64);

  // ---- KF-dispatched GEMM launch ----
  auto launch_gemm = [&](const _Float16* A, const _Float16* W, _Float16* out,
                         int kdim, int cout) {
    int waves = (RTOT / 64) * (cout / 16);   // 4 row tiles per wave
    dim3 grid((waves + 7) / 8), blk(256);
    switch (kdim >> 5) {
      case 1: flot_gemm_wmma<1><<<grid, blk, 0, stream>>>(A, W, out, RTOT, cout); break;
      case 2: flot_gemm_wmma<2><<<grid, blk, 0, stream>>>(A, W, out, RTOT, cout); break;
      case 3: flot_gemm_wmma<3><<<grid, blk, 0, stream>>>(A, W, out, RTOT, cout); break;
      default: flot_gemm_wmma<4><<<grid, blk, 0, stream>>>(A, W, out, RTOT, cout); break;
    }
  };

  // ---- one 3-conv set_conv block ----
  auto launch_setconv = [&](const float* x, int xstride, int cin, int base, int cout,
                            const int* nbr, const float* pc, float* outFeats) {
    int cin0 = cin + 3;
    int cin0pad = (cin0 + 31) & ~31;
    flot_gather<<<dim3((RTOT + 255) / 256), dim3(256), 0, stream>>>(
        x, xstride, cin, pc, nbr, h_buf, cin0pad, N);

    auto conv = [&](const _Float16* A, int kdim, const float* W, int wcin,
                    const float* g, const float* bb, _Float16* out) {
      flot_repack_w<<<dim3((cout * kdim + 255) / 256), dim3(256), 0, stream>>>(
          W, wbuf, cout, wcin, kdim);
      launch_gemm(A, wbuf, out, kdim, cout);
      flot_in_stats<<<dim3(B_SZ * cout), dim3(256), 0, stream>>>(out, stats, cout);
      size_t tot = (size_t)RTOT * cout;
      flot_in_apply<<<dim3((unsigned)((tot + 255) / 256)), dim3(256), 0, stream>>>(
          out, stats, g, bb, cout);
    };
    const float* W0 = (const float*)d_in[base + 0];
    const float* W1 = (const float*)d_in[base + 1];
    const float* W2 = (const float*)d_in[base + 2];
    const float* b0 = (const float*)d_in[base + 3];
    const float* b1 = (const float*)d_in[base + 4];
    const float* b2 = (const float*)d_in[base + 5];
    const float* g0 = (const float*)d_in[base + 6];
    const float* g1 = (const float*)d_in[base + 7];
    const float* g2 = (const float*)d_in[base + 8];

    conv(h_buf, cin0pad, W0, cin0, g0, b0, conv_buf);
    conv(conv_buf, cout, W1, cout, g1, b1, h_buf);
    conv(h_buf, cout, W2, cout, g2, b2, conv_buf);
    flot_maxpool<<<dim3((B_SZ * N * cout + 255) / 256), dim3(256), 0, stream>>>(
        conv_buf, outFeats, cout, N);
  };

  // ---- full 3-set_conv stack; result lands in feats_a (128 ch) ----
  auto run_stack = [&](const float* pcX, const int* nbr, int base,
                       const float* xinit, int xstride, int xcin) {
    launch_setconv(xinit, xstride, xcin, base + 0, 32, nbr, pcX, feats_a);
    launch_setconv(feats_a, LDH, 32, base + 9, 64, nbr, pcX, feats_b);
    launch_setconv(feats_b, LDH, 64, base + 18, 128, nbr, pcX, feats_a);
  };

  // ================= pipeline =================
  flot_scalars<<<dim3(1), dim3(32), 0, stream>>>(epsP, gamP, scalars);
  flot_sqnorm<<<dim3((B_SZ * N + 255) / 256), dim3(256), 0, stream>>>(pc1, sq1, B_SZ * N);
  flot_sqnorm<<<dim3((B_SZ * N + 255) / 256), dim3(256), 0, stream>>>(pc2, sq2, B_SZ * N);
  flot_knn<<<dim3(N / 256, B_SZ), dim3(256), 0, stream>>>(pc1, nbr1, N);
  flot_knn<<<dim3(N / 256, B_SZ), dim3(256), 0, stream>>>(pc2, nbr2, N);

  run_stack(pc1, nbr1, featBase, pc1, 3, 3);
  flot_featnorm<<<dim3(B_SZ * N), dim3(128), 0, stream>>>(feats_a, fnorm1);
  run_stack(pc2, nbr2, featBase, pc2, 3, 3);
  flot_featnorm<<<dim3(B_SZ * N), dim3(128), 0, stream>>>(feats_a, fnorm2);

  flot_colsum<<<dim3(B_SZ * (N / 16)), dim3(128), 0, stream>>>(
      fnorm1, fnorm2, pc1, pc2, sq1, sq2, scalars, colsum, N);
  flot_bvec<<<dim3((B_SZ * N + 255) / 256), dim3(256), 0, stream>>>(colsum, scalars, bvec, N);
  flot_rowpass<<<dim3(B_SZ * (N / 16)), dim3(128), 0, stream>>>(
      fnorm1, fnorm2, pc1, pc2, sq1, sq2, bvec, scalars, rowacc, N);
  flot_otflow<<<dim3((B_SZ * N + 255) / 256), dim3(256), 0, stream>>>(
      rowacc, scalars, pc1, otflow, N);

  run_stack(pc1, nbr1, refBase, otflow, 3, 3);
  flot_fc<<<dim3((B_SZ * N + 255) / 256), dim3(256), 0, stream>>>(
      feats_a, fcw, fcb, otflow, (float*)d_out, B_SZ * N);
}
